// ScaledDotProductAttention_28578712387681
// MI455X (gfx1250) — matrix operand order repaired and checked by reading, not compiled, not run on hardware
//
#include <hip/hip_runtime.h>
#include <hip/hip_bf16.h>
#include <cstdint>
#include <cstddef>

// ---------------------------------------------------------------------------
// Scaled dot-product attention, MI455X (gfx1250), wave32, WMMA bf16 path.
// B=4 H=8 T=2048 DK=64. Outputs: [output (B,H,T,DK) | attn_weights (B,H,T,T)].
// Memory-bound design: ~740 MB moved (537 MB weight stream written NT),
// ~34 GFLOP pushed through v_wmma_f32_16x16x32_bf16.
// ---------------------------------------------------------------------------

#define T_SEQ   2048
#define DK_DIM  64
#define H_HEADS 8
#define B_BATCH 4
#define ROWPAD  2052          // 2048 + 4 floats: row stride % 64 banks == 4

typedef __attribute__((ext_vector_type(16))) __bf16 v16bf;
typedef __attribute__((ext_vector_type(8)))  float  v8f;
typedef __attribute__((ext_vector_type(4)))  float  v4f;   // native vec for NT stores

// Write a packed bf16 pair directly into the fragment: lets the compiler use
// the native packed f32->bf16 convert instead of a manual RNE bit sequence.
__device__ __forceinline__ void set2(v16bf& f, int i, float lo, float hi) {
    f[2 * i]     = (__bf16)lo;
    f[2 * i + 1] = (__bf16)hi;
}

__global__ __launch_bounds__(256)
void sdpa_wmma_kernel(const float* __restrict__ Q, const float* __restrict__ K,
                      const float* __restrict__ V, const float* __restrict__ bias,
                      float* __restrict__ out, float* __restrict__ wts) {
    extern __shared__ float smem[];
    float* sS    = smem;                    // [16][ROWPAD] score strip (f32)
    float* sPart = smem + 16 * ROWPAD;      // [4][256] partial O accumulators

    const int tid  = threadIdx.x;
    const int wave = tid >> 5;              // 8 waves
    const int lane = tid & 31;
    const int lh   = lane >> 4;             // lane half (0/1)
    const int ln   = lane & 15;

    const int qt    = blockIdx.x & (T_SEQ / 16 - 1);   // 128 q-tiles
    const int bh    = blockIdx.x >> 7;                 // b*H + h  (0..31)
    const int h     = bh & (H_HEADS - 1);
    const int qbase = qt * 16;

    const float* __restrict__ Qp = Q + (size_t)bh * T_SEQ * DK_DIM;
    const float* __restrict__ Kp = K + (size_t)bh * T_SEQ * DK_DIM;
    const float* __restrict__ Vp = V + (size_t)bh * T_SEQ * DK_DIM;
    const float* __restrict__ Bp = bias + (size_t)h * T_SEQ * T_SEQ;
    float* __restrict__ Op = out + (size_t)bh * T_SEQ * DK_DIM;
    float* __restrict__ Wp = wts + (size_t)bh * T_SEQ * T_SEQ;

    const float scale = 0.125f;             // 1/sqrt(64)

    // ---- Q A-fragments (per-wave private copy; reused over this wave's 16 k-tiles)
    // A layout (16-bit 16x32): lane m = ln; VGPR v holds
    //   k = (v>>2)*16 + lh*8 + (v&3)*2  (+1 in high half)
    v16bf aQ0, aQ1;
    {
        const float* qrow = Qp + (size_t)(qbase + ln) * DK_DIM;
        #pragma unroll
        for (int v = 0; v < 8; ++v) {
            const int k0 = ((v >> 2) << 4) + (lh << 3) + ((v & 3) << 1);
            const float2 q0 = *(const float2*)(qrow + k0);
            const float2 q1 = *(const float2*)(qrow + 32 + k0);
            set2(aQ0, v, q0.x, q0.y);
            set2(aQ1, v, q1.x, q1.y);
        }
    }

    // =========================== Phase 1: scores ===========================
    // S[m][n] = scale * sum_k Q[qbase+m][k] * K[kbase+n][k] + bias
    // B layout (32x16): lane n = ln; VGPR v holds k = (v>>2)*16 + lh*8 + (v&3)*2 (+1 hi).
    for (int kt = wave; kt < T_SEQ / 16; kt += 8) {
        const int kbase = kt * 16;
        v16bf bK0, bK1;
        const float* krow = Kp + (size_t)(kbase + ln) * DK_DIM;
        #pragma unroll
        for (int v = 0; v < 8; ++v) {
            const int k0 = ((v >> 2) << 4) + (lh << 3) + ((v & 3) << 1);
            const float2 kk0 = *(const float2*)(krow + k0);
            const float2 kk1 = *(const float2*)(krow + 32 + k0);
            set2(bK0, v, kk0.x, kk0.y);
            set2(bK1, v, kk1.x, kk1.y);
        }
        v8f c = {};
        c = __builtin_amdgcn_wmma_f32_16x16x32_bf16(false, aQ0, false, bK0,
                                                    (short)0, c, false, false);
        c = __builtin_amdgcn_wmma_f32_16x16x32_bf16(false, aQ1, false, bK1,
                                                    (short)0, c, false, false);
        // C/D layout: lane ln holds column n=ln; VGPR v is row m = v + 8*lh
        #pragma unroll
        for (int v = 0; v < 8; ++v) {
            const int m   = v + (lh << 3);
            const int col = kbase + ln;
            const float s = c[v] * scale + Bp[(size_t)(qbase + m) * T_SEQ + col];
            sS[m * ROWPAD + col] = s;
        }
    }
    __syncthreads();

    // =========================== Phase 2: softmax ==========================
    // 16 threads per row; v4f sweeps; shuffle-reduce within 16 lanes.
    // Single exp evaluation: exp(x-mx) is written back to LDS in the sum pass.
    {
        const int r   = tid >> 4;           // row 0..15
        const int c16 = tid & 15;
        float* row = sS + r * ROWPAD;

        float mx = -3.402823466e38f;
        #pragma unroll 8
        for (int j = 0; j < 32; ++j) {
            const v4f x = *(const v4f*)(row + ((c16 + j * 16) << 2));
            mx = fmaxf(fmaxf(mx, fmaxf(x.x, x.y)), fmaxf(x.z, x.w));
        }
        #pragma unroll
        for (int off = 8; off; off >>= 1) mx = fmaxf(mx, __shfl_xor(mx, off, 16));

        float sum = 0.f;
        #pragma unroll 4
        for (int j = 0; j < 32; ++j) {
            const int idx = (c16 + j * 16) << 2;
            v4f x = *(v4f*)(row + idx);
            x.x = __expf(x.x - mx);
            x.y = __expf(x.y - mx);
            x.z = __expf(x.z - mx);
            x.w = __expf(x.w - mx);
            *(v4f*)(row + idx) = x;          // stash exp for the normalize pass
            sum += (x.x + x.y) + (x.z + x.w);
        }
        #pragma unroll
        for (int off = 8; off; off >>= 1) sum += __shfl_xor(sum, off, 16);
        const float inv = 1.0f / sum;

        float* grow = Wp + (size_t)(qbase + r) * T_SEQ;
        #pragma unroll 4
        for (int j = 0; j < 32; ++j) {
            const int idx = (c16 + j * 16) << 2;
            v4f x = *(v4f*)(row + idx);
            x.x *= inv; x.y *= inv; x.z *= inv; x.w *= inv;
            *(v4f*)(row + idx) = x;                              // keep for GEMM2
            __builtin_nontemporal_store(x, (v4f*)(grow + idx));  // 537MB stream
        }
    }
    __syncthreads();

    // ========================= Phase 3: O = W @ V ==========================
    // 8 waves = 4 n-tiles x 2 k-halves of 1024. 32 WMMAs per wave.
    {
        const int ntile   = wave & 3;
        const int kh      = wave >> 2;
        const int colbase = ntile * 16;
        v8f c = {};
        for (int kt = 0; kt < 32; ++kt) {
            const int kbase = kh * 1024 + kt * 32;
            v16bf aW, bV;
            const float* srow = sS + ln * ROWPAD + kbase;       // A: m = ln
            #pragma unroll
            for (int v = 0; v < 8; ++v) {
                const int k0 = ((v >> 2) << 4) + (lh << 3) + ((v & 3) << 1);
                const float2 w2 = *(const float2*)(srow + k0);
                set2(aW, v, w2.x, w2.y);
                const int kb = kbase + ((v >> 2) << 4) + (lh << 3) + ((v & 3) << 1);    // B: k same as A
                set2(bV, v, Vp[(size_t)kb * DK_DIM + colbase + ln],
                            Vp[(size_t)(kb + 1) * DK_DIM + colbase + ln]);
            }
            c = __builtin_amdgcn_wmma_f32_16x16x32_bf16(false, aW, false, bV,
                                                        (short)0, c, false, false);
        }
        if (kh == 1) {
            #pragma unroll
            for (int v = 0; v < 8; ++v)
                sPart[ntile * 256 + lane * 8 + v] = c[v];
        }
        __syncthreads();
        if (kh == 0) {
            #pragma unroll
            for (int v = 0; v < 8; ++v) {
                const float o = c[v] + sPart[ntile * 256 + lane * 8 + v];
                const int m = v + (lh << 3);
                Op[(size_t)(qbase + m) * DK_DIM + colbase + ln] = o;
            }
        }
    }
}

extern "C" void kernel_launch(void* const* d_in, const int* in_sizes, int n_in,
                              void* d_out, int out_size, void* d_ws, size_t ws_size,
                              hipStream_t stream) {
    (void)in_sizes; (void)n_in; (void)out_size; (void)d_ws; (void)ws_size;
    const float* Q    = (const float*)d_in[0];
    const float* K    = (const float*)d_in[1];
    const float* V    = (const float*)d_in[2];
    const float* bias = (const float*)d_in[3];
    float* out = (float*)d_out;
    float* wts = out + (size_t)B_BATCH * H_HEADS * T_SEQ * DK_DIM;

    const size_t shmem = (size_t)(16 * ROWPAD + 4 * 256) * sizeof(float); // ~135 KB
    // gfx1250 WGP has 320 KB LDS; raise the dynamic-LDS cap (deterministic,
    // not a stream op — safe under graph capture).
    (void)hipFuncSetAttribute((const void*)sdpa_wmma_kernel,
                              hipFuncAttributeMaxDynamicSharedMemorySize,
                              (int)shmem);

    dim3 grid(B_BATCH * H_HEADS * (T_SEQ / 16));   // 4096 workgroups
    dim3 block(256);                               // 8 waves (wave32)
    sdpa_wmma_kernel<<<grid, block, shmem, stream>>>(Q, K, V, bias, out, wts);
}
